// ilt_loss_layer_75041668596436
// MI455X (gfx1250) — matrix operand order repaired and checked by reading, not compiled, not run on hardware
//
#include <hip/hip_runtime.h>
#include <hip/hip_bf16.h>

typedef __attribute__((ext_vector_type(16))) _Float16 v16h;
typedef __attribute__((ext_vector_type(8)))  float    v8f;
typedef __attribute__((ext_vector_type(4)))  unsigned int v4u;
typedef __attribute__((ext_vector_type(8)))  int      v8i;
typedef __attribute__((ext_vector_type(4)))  int      v4i;

#define HH     1024
#define WW     1024
#define KK     24
#define KSZ    35
#define BB     2
#define NCH    48            // 2*KK channels (re/im)
#define NT     3             // N tiles of 16
#define CHROW  5             // 8-wide chunks per kernel row (40 padded cols)
#define NCHUNK (KSZ*CHROW)   // 175
#define NSTEP  44            // ceil(175/4) K=32 WMMA steps
#define TILE_W 256
#define LDSW   (TILE_W + 40) // halo: 296
#define NWGX   (WW / TILE_W) // 4
#define NWG    (BB * HH * NWGX) // 8192

// ds_swizzle group-of-32 XOR pattern: {and=0x1f, or=0, xor=mask}
#define SWZ(m) (((m) << 10) | 0x1f)

__device__ __forceinline__ float xor_add(float v, int pat_hi, int m) {
    (void)pat_hi;
    int r;
    switch (m) {  // immediate-offset forms, fully unrolled callers
        case 1:  r = __builtin_amdgcn_ds_swizzle(__float_as_int(v), SWZ(1));  break;
        case 2:  r = __builtin_amdgcn_ds_swizzle(__float_as_int(v), SWZ(2));  break;
        case 4:  r = __builtin_amdgcn_ds_swizzle(__float_as_int(v), SWZ(4));  break;
        case 8:  r = __builtin_amdgcn_ds_swizzle(__float_as_int(v), SWZ(8));  break;
        default: r = __builtin_amdgcn_ds_swizzle(__float_as_int(v), SWZ(16)); break;
    }
    return v + __int_as_float(r);
}

// chunk id -> LDS element offset (row*LDSW + col0); folds to an immediate
// once the K loop is fully unrolled.
constexpr int coff(int c) {
    int r  = c / CHROW;
    int j0 = (c - r * CHROW) * 8;
    if (r > KSZ - 1) r = KSZ - 1;    // padding chunk: any valid row, B is zero there
    return r * LDSW + j0;
}

// ---------------- prep 1: mask sigmoid -> fp16 ----------------
__global__ void k_masksig(const float* __restrict__ m, _Float16* __restrict__ out, int n) {
    int i = blockIdx.x * blockDim.x + threadIdx.x;
    if (i < n) {
        float two = m[i] * 2.0f - 1.0f;
        float s = 1.0f / (1.0f + __expf(-4.0f * two));   // THETA_M = 4
        out[i] = (_Float16)s;
    }
}

// ---------------- prep 2: repack kernels into per-lane WMMA B records ----------------
// Layout: bp[((s*NT + f)*32 + lane)*16 + h], fp16. Lane l<16: N=l, K=0..15; l>=16: N=l-16, K=16..31.
__global__ void k_repack(const float* __restrict__ kern, const float* __restrict__ wgt,
                         _Float16* __restrict__ bp) {
    int i = blockIdx.x * blockDim.x + threadIdx.x;
    const int total = NSTEP * NT * 32 * 16;
    if (i >= total) return;
    int h    = i & 15;
    int rest = i >> 4;
    int l    = rest & 31;
    int sf   = rest >> 5;
    int f    = sf % NT;
    int s    = sf / NT;
    int n    = f * 16 + (l & 15);        // channel 0..47
    int k    = n >> 1;                   // SOCS kernel index
    int part = n & 1;                    // 0=real, 1=imag
    int klocal = ((l >> 4) << 4) + h;    // 0..31
    int chunk  = 4 * s + (klocal >> 3);
    int jj     = klocal & 7;
    float v = 0.0f;
    if (chunk < NCHUNK) {
        int r  = chunk / CHROW;
        int c  = (chunk % CHROW) * 8 + jj;
        if (c < KSZ)
            v = kern[(((k * KSZ) + r) * KSZ + c) * 2 + part] * sqrtf(wgt[k]);
    }
    bp[i] = (_Float16)v;
}

// ---------------- main: implicit-GEMM convolution + intensity + loss partials ----------------
__global__ __launch_bounds__(256)
void k_conv(const _Float16* __restrict__ msig, const _Float16* __restrict__ bpack,
            const float* __restrict__ target, float* __restrict__ partials) {
    __shared__ _Float16 smask[KSZ * LDSW];
    __shared__ float    sred[8];

    const int bid = blockIdx.x;
    const int cx  = bid & (NWGX - 1);
    const int y   = (bid >> 2) & (HH - 1);
    const int b   = bid >> 12;           // 2 + 10 bits
    const int x0  = cx * TILE_W;
    const int tid = threadIdx.x;

    // ---- stage 35x296 fp16 halo tile of mask_sig into LDS ----
    // Interior tiles (no circular wrap) use the Tensor Data Mover: one wave issues
    // a 2D D# (tile 296x35, 2B elements, row stride 1024) and waits on TENSORcnt.
    const bool interior = (y >= KSZ / 2) && (y <= HH - 1 - KSZ / 2) &&
                          (cx >= 1) && (cx <= NWGX - 2);
    if (interior) {
        if (tid < 32) {
            const _Float16* gsrc =
                msig + (((size_t)((b << 10) + (y - KSZ / 2))) << 10) + (x0 - KSZ / 2);
            unsigned long long ga = (unsigned long long)(uintptr_t)gsrc;
            unsigned int lp = (unsigned int)(uintptr_t)&smask[0];   // LDS byte offset (addr[31:0])
            v4u g0 = { 1u,                                   // count=1 (valid), user mode
                       lp,                                   // lds_addr
                       (unsigned int)ga,                     // global_addr[31:0]
                       (unsigned int)(ga >> 32) | 0x80000000u }; // addr[56:32] | type=2<<30
            v8i g1 = { 0x00010000,                           // data_size=1 (2 bytes), no flags
                       0,                                    // barrier addr / tensor_dim0 lo16=0
                       0x4000,                               // tensor_dim0=1<<30 (no clip)
                       (int)(0x4000u | (296u << 16)),        // tensor_dim1=1<<30, tile_dim0=296
                       35,                                   // tile_dim1=35, tile_dim2=0
                       1024,                                 // tensor_dim0_stride=1024 elements
                       0, 0 };
            v4i z4 = { 0, 0, 0, 0 };
#if __clang_major__ >= 23
            v8i z8 = { 0, 0, 0, 0, 0, 0, 0, 0 };
            __builtin_amdgcn_tensor_load_to_lds(g0, g1, z4, z4, z8, 0);
#else
            __builtin_amdgcn_tensor_load_to_lds(g0, g1, z4, z4, 0);
#endif
            __builtin_amdgcn_s_wait_tensorcnt(0);
        }
        __syncthreads();
    } else {
        // border tile: manual staging with circular wrap (H, W are powers of two)
        for (int r = 0; r < KSZ; ++r) {
            int gy = (y + r - KSZ / 2 + HH) & (HH - 1);
            const _Float16* srow = msig + (((size_t)((b << 10) + gy)) << 10);
            for (int c = tid; c < LDSW; c += 256) {
                int gx = (x0 + c - KSZ / 2 + WW) & (WW - 1);
                smask[r * LDSW + c] = srow[gx];
            }
        }
        __syncthreads();
    }

    const int lane = tid & 31;
    const int wave = tid >> 5;
    const int grp  = lane >> 4;          // A/B half-wave group
    const int lm   = lane & 15;

    // Two 16-pixel M tiles per wave share every B fragment (K-step outer loop).
    const _Float16* baseA0 = smask + (wave << 5) + lm;        // mtile 2*wave
    const _Float16* baseA1 = baseA0 + 16;                     // mtile 2*wave+1
    const v16h*     bbase  = (const v16h*)bpack + lane;       // 32B record per lane

    v8f acc[2][3] = {};

    #pragma unroll
    for (int s = 0; s < NSTEP; ++s) {
        // A-fragment K layout: lanes 0-15 take chunks 4s+0 / 4s+1 (K 0-7 / 8-15),
        // lanes 16-31 take chunks 4s+2 / 4s+3 (K 16-23 / 24-31). With s constant
        // after unroll, coff() folds; grp select becomes a cndmask pair.
        const int oLo = grp ? coff(4 * s + 2) : coff(4 * s + 0);
        const int oHi = grp ? coff(4 * s + 3) : coff(4 * s + 1);

        // B fragments: shared by both M tiles; immediate offsets off one base.
        v16h b0 = bbase[(s * NT + 0) * 32];
        v16h b1 = bbase[(s * NT + 1) * 32];
        v16h b2 = bbase[(s * NT + 2) * 32];
        __builtin_prefetch(bbase + (s * NT + 3) * 32, 0, 3);

        {   // M tile 0
            const _Float16* pLo = baseA0 + oLo;
            const _Float16* pHi = baseA0 + oHi;
            v16h a;
            #pragma unroll
            for (int h = 0; h < 8; ++h) { a[h] = pLo[h]; a[h + 8] = pHi[h]; }
            acc[0][0] = __builtin_amdgcn_wmma_f32_16x16x32_f16(false, a, false, b0, (short)0, acc[0][0], false, false);
            acc[0][1] = __builtin_amdgcn_wmma_f32_16x16x32_f16(false, a, false, b1, (short)0, acc[0][1], false, false);
            acc[0][2] = __builtin_amdgcn_wmma_f32_16x16x32_f16(false, a, false, b2, (short)0, acc[0][2], false, false);
        }
        {   // M tile 1
            const _Float16* pLo = baseA1 + oLo;
            const _Float16* pHi = baseA1 + oHi;
            v16h a;
            #pragma unroll
            for (int h = 0; h < 8; ++h) { a[h] = pLo[h]; a[h + 8] = pHi[h]; }
            acc[1][0] = __builtin_amdgcn_wmma_f32_16x16x32_f16(false, a, false, b0, (short)0, acc[1][0], false, false);
            acc[1][1] = __builtin_amdgcn_wmma_f32_16x16x32_f16(false, a, false, b1, (short)0, acc[1][1], false, false);
            acc[1][2] = __builtin_amdgcn_wmma_f32_16x16x32_f16(false, a, false, b2, (short)0, acc[1][2], false, false);
        }
    }

    // Intensity per pixel: sum field^2 over 48 channels.
    // C layout: VGPR g, lanes 0-15 -> pixel M=g, lanes 16-31 -> M=g+8; N = lm (+16 per fragment).
    float lacc = 0.0f;
    #pragma unroll
    for (int mt = 0; mt < 2; ++mt) {
        const int mbase = (wave << 5) + (mt << 4);
        #pragma unroll
        for (int g = 0; g < 8; ++g) {
            float t = acc[mt][0][g] * acc[mt][0][g]
                    + acc[mt][1][g] * acc[mt][1][g]
                    + acc[mt][2][g] * acc[mt][2][g];
            t = xor_add(t, 0, 1);
            t = xor_add(t, 0, 2);
            t = xor_add(t, 0, 4);
            t = xor_add(t, 0, 8);        // reduce over the 16 N-lanes of this half
            if (lm == g) {
                int px = x0 + mbase + g + (grp << 3);
                float tgt   = target[((b << 10) + y) * WW + px];
                float wafer = 1.0f / (1.0f + __expf(-50.0f * (t - 0.225f)));  // THETA_Z, TH
                float d  = wafer - tgt;
                float d2 = d * d;
                lacc += d2 * d2;          // GAMMA = 4
            }
        }
    }

    // wave + block reduction -> per-block partial (deterministic, no atomics)
    lacc = xor_add(lacc, 0, 16);
    lacc = xor_add(lacc, 0, 8);
    lacc = xor_add(lacc, 0, 4);
    lacc = xor_add(lacc, 0, 2);
    lacc = xor_add(lacc, 0, 1);
    if (lane == 0) sred[wave] = lacc;
    __syncthreads();
    if (tid == 0) {
        float s = 0.0f;
        #pragma unroll
        for (int i = 0; i < 8; ++i) s += sred[i];
        partials[bid] = s;
    }
}

// ---------------- final deterministic reduction ----------------
__global__ void k_reduce(const float* __restrict__ partials, float* __restrict__ out, int n) {
    __shared__ float sred[256];
    int tid = threadIdx.x;
    float s = 0.0f;
    for (int i = tid; i < n; i += 256) s += partials[i];
    sred[tid] = s;
    __syncthreads();
    for (int off = 128; off > 0; off >>= 1) {
        if (tid < off) sred[tid] += sred[tid + off];
        __syncthreads();
    }
    if (tid == 0) out[0] = sred[0];
}

extern "C" void kernel_launch(void* const* d_in, const int* in_sizes, int n_in,
                              void* d_out, int out_size, void* d_ws, size_t ws_size,
                              hipStream_t stream) {
    const float* mask_pred = (const float*)d_in[0];   // [B,1,H,W]
    const float* target    = (const float*)d_in[1];   // [B,1,H,W]
    const float* kernels   = (const float*)d_in[2];   // [K,KS,KS,2]
    const float* weight    = (const float*)d_in[3];   // [K]
    float* out = (float*)d_out;

    // workspace layout
    _Float16* msig  = (_Float16*)d_ws;                          // 2M halves = 4 MB
    _Float16* bpack = msig + (size_t)BB * HH * WW;              // 67584 halves (32B-aligned offset)
    float*    partials = (float*)(bpack + (size_t)NSTEP * NT * 32 * 16);

    const int nmask = BB * HH * WW;
    k_masksig<<<(nmask + 255) / 256, 256, 0, stream>>>(mask_pred, msig, nmask);

    const int nbp = NSTEP * NT * 32 * 16;
    k_repack<<<(nbp + 255) / 256, 256, 0, stream>>>(kernels, weight, bpack);

    k_conv<<<NWG, 256, 0, stream>>>(msig, bpack, target, partials);

    k_reduce<<<1, 256, 0, stream>>>(partials, out, NWG);
}
